// GATEncoder_14705968022274
// MI455X (gfx1250) — matrix operand order repaired and checked by reading, not compiled, not run on hardware
//
#include <hip/hip_runtime.h>
#include <hip/hip_bf16.h>

#define NNODES 50000
#define NEDGES 800000
#define ETOT   (NNODES + NEDGES)
#define FIN    256
#define HC     256
#define NHEADS 4
#define CDIM   64
#define NEG_SLOPE 0.2f

typedef __attribute__((ext_vector_type(16))) __bf16 v16bf;
typedef __attribute__((ext_vector_type(8)))  float  v8f;

union BF16Frag { v16bf v; unsigned int u[8]; };

__device__ __forceinline__ unsigned int bf16_hi_bits(float f) {
  return __builtin_bit_cast(unsigned int, f) >> 16;
}
__device__ __forceinline__ float bf16_res(float f) {
  unsigned int b = __builtin_bit_cast(unsigned int, f) & 0xFFFF0000u;
  return f - __builtin_bit_cast(float, b);
}

// order-preserving float<->uint key for atomicMax (key 0 == -inf sentinel)
__device__ __forceinline__ unsigned int fkey(float f) {
  unsigned int b = __builtin_bit_cast(unsigned int, f);
  return (b & 0x80000000u) ? ~b : (b | 0x80000000u);
}
__device__ __forceinline__ float fkey_dec(unsigned int k) {
  unsigned int b = (k & 0x80000000u) ? (k & 0x7FFFFFFFu) : ~k;
  return __builtin_bit_cast(float, b);
}

__device__ __forceinline__ void edge_nodes(const int* __restrict__ ei,
                                           int e, int& s, int& d) {
  if (e < NEDGES) { s = ei[e]; d = ei[NEDGES + e]; }
  else { s = e - NEDGES; d = s; }   // appended self-loops
}

// ---------------------------------------------------------------- utilities
__global__ void zero_kernel(float* __restrict__ p, int n) {
  int i = blockIdx.x * blockDim.x + threadIdx.x;
  if (i < n) p[i] = 0.0f;
}

// W [k][n] f32  ->  Wt_hi/Wt_lo [n][k] bf16 (split hi/lo)
__global__ void wprep_kernel(const float* __restrict__ W,
                             unsigned short* __restrict__ wt_hi,
                             unsigned short* __restrict__ wt_lo) {
  int idx = blockIdx.x * blockDim.x + threadIdx.x;
  if (idx >= FIN * HC) return;
  int n = idx / FIN, k = idx % FIN;
  float f = W[k * HC + n];
  wt_hi[idx] = (unsigned short)bf16_hi_bits(f);
  wt_lo[idx] = (unsigned short)bf16_hi_bits(bf16_res(f));
}

// ------------------------------------------------- h = x @ W via bf16x3 WMMA
// block = 256 thr = 8 waves; wave w: rows [blk*32 + 16*(w>>2) .. +15],
// head (w&3) -> cols [64*head .. 64*head+63] as 4 WMMA 16x16 tiles. K=256.
__global__ void __launch_bounds__(256)
gemm_bf16x3_kernel(const float* __restrict__ x,
                   const unsigned short* __restrict__ wt_hi,
                   const unsigned short* __restrict__ wt_lo,
                   float* __restrict__ hbuf) {
  const int lane   = threadIdx.x & 31;
  const int wave   = threadIdx.x >> 5;
  const int laneN  = lane & 15;
  const int laneHi = lane >> 4;
  const int row0   = blockIdx.x * 32 + (wave >> 2) * 16;
  const int col0   = (wave & 3) * 64;

  int rowA = row0 + laneN;
  if (rowA >= NNODES) rowA = NNODES - 1;
  const float* __restrict__ xrow = x + (long)rowA * FIN;

  v8f acc[4];
  #pragma unroll
  for (int t = 0; t < 4; ++t)
    #pragma unroll
    for (int r = 0; r < 8; ++r) acc[t][r] = 0.0f;

  for (int k0 = 0; k0 < FIN; k0 += 32) {
    // ---- A fragments (16x32 bf16 lane layout), f32 -> hi/lo split on the fly
    BF16Frag a_hi, a_lo;
    #pragma unroll
    for (int p = 0; p < 8; ++p) {
      int k = k0 + ((p < 4) ? 0 : 16) + laneHi * 8 + 2 * (p & 3);
      float f0 = xrow[k], f1 = xrow[k + 1];
      a_hi.u[p] = bf16_hi_bits(f0) | (bf16_hi_bits(f1) << 16);
      a_lo.u[p] = bf16_hi_bits(bf16_res(f0)) |
                  (bf16_hi_bits(bf16_res(f1)) << 16);
    }
    // ---- B fragments (32x16 bf16) from transposed bf16 W, packed dword loads
    #pragma unroll
    for (int t = 0; t < 4; ++t) {
      int ncol = col0 + t * 16 + laneN;
      const unsigned int* __restrict__ bh =
          (const unsigned int*)(wt_hi + (long)ncol * FIN);
      const unsigned int* __restrict__ bl =
          (const unsigned int*)(wt_lo + (long)ncol * FIN);
      BF16Frag b_hi, b_lo;
      #pragma unroll
      for (int p = 0; p < 8; ++p) {
        int kk = k0 + ((p < 4) ? 0 : 16) + laneHi * 8 + 2 * (p & 3);     // element K = 8*laneHi + e (e<8), 16 + 8*laneHi + (e-8) (e>=8)
        b_hi.u[p] = bh[kk >> 1];
        b_lo.u[p] = bl[kk >> 1];
      }
      acc[t] = __builtin_amdgcn_wmma_f32_16x16x32_bf16(
          false, a_hi.v, false, b_hi.v, (short)0, acc[t], false, false);
      acc[t] = __builtin_amdgcn_wmma_f32_16x16x32_bf16(
          false, a_hi.v, false, b_lo.v, (short)0, acc[t], false, false);
      acc[t] = __builtin_amdgcn_wmma_f32_16x16x32_bf16(
          false, a_lo.v, false, b_hi.v, (short)0, acc[t], false, false);
    }
  }
  // ---- store C/D: VGPR r, lane -> M = r + 8*laneHi, N = laneN
  #pragma unroll
  for (int t = 0; t < 4; ++t)
    #pragma unroll
    for (int r = 0; r < 8; ++r) {
      int row = row0 + r + 8 * laneHi;
      if (row < NNODES)
        hbuf[(long)row * HC + col0 + t * 16 + laneN] = acc[t][r];
    }
}

// ------------------------------------------------------ per-node attn coeffs
__global__ void att_dot_kernel(const float* __restrict__ h,
                               const float* __restrict__ att_src,
                               const float* __restrict__ att_dst,
                               float* __restrict__ asrc,
                               float* __restrict__ adst) {
  int idx = blockIdx.x * blockDim.x + threadIdx.x;   // n*NHEADS + head
  if (idx >= NNODES * NHEADS) return;
  int n = idx >> 2, head = idx & 3;
  const float4* hv = (const float4*)(h + (long)n * HC + head * CDIM);
  const float4* as = (const float4*)(att_src + head * CDIM);
  const float4* ad = (const float4*)(att_dst + head * CDIM);
  float s = 0.0f, d = 0.0f;
  #pragma unroll
  for (int i = 0; i < 16; ++i) {
    float4 v = hv[i], a = as[i], b = ad[i];
    s += v.x * a.x + v.y * a.y + v.z * a.z + v.w * a.w;
    d += v.x * b.x + v.y * b.y + v.z * b.z + v.w * b.w;
  }
  asrc[idx] = s; adst[idx] = d;
}

// --------------------------------------------------- pass 1: segment max (e)
__global__ void edge_max_kernel(const int* __restrict__ ei,
                                const float* __restrict__ asrc,
                                const float* __restrict__ adst,
                                unsigned int* __restrict__ mkey) {
  int idx = blockIdx.x * blockDim.x + threadIdx.x;
  if (idx >= ETOT * NHEADS) return;
  int e = idx >> 2, head = idx & 3;
  int s, d; edge_nodes(ei, e, s, d);
  float v = asrc[s * 4 + head] + adst[d * 4 + head];
  v = (v >= 0.0f) ? v : NEG_SLOPE * v;
  atomicMax(mkey + d * 4 + head, fkey(v));
}

__global__ void decode_max_kernel(const unsigned int* __restrict__ mkey,
                                  float* __restrict__ mflt) {
  int i = blockIdx.x * blockDim.x + threadIdx.x;
  if (i < NNODES * NHEADS) mflt[i] = fkey_dec(mkey[i]);
}

// -------------------------------------------------- pass 2: softmax denom sum
__global__ void edge_den_kernel(const int* __restrict__ ei,
                                const float* __restrict__ asrc,
                                const float* __restrict__ adst,
                                const float* __restrict__ m,
                                float* __restrict__ denom) {
  int idx = blockIdx.x * blockDim.x + threadIdx.x;
  if (idx >= ETOT * NHEADS) return;
  int e = idx >> 2, head = idx & 3;
  int s, d; edge_nodes(ei, e, s, d);
  int d4 = d * 4 + head;
  float v = asrc[s * 4 + head] + adst[d4];
  v = (v >= 0.0f) ? v : NEG_SLOPE * v;
  atomicAdd(denom + d4, expf(v - m[d4]));
}

__global__ void inv_den_kernel(float* __restrict__ denom) {
  int i = blockIdx.x * blockDim.x + threadIdx.x;
  if (i < NNODES * NHEADS) denom[i] = 1.0f / (denom[i] + 1e-16f);
}

// ---------------------------------- pass 3: one wave per edge, 8 ch per lane
__global__ void __launch_bounds__(256)
edge_agg_kernel(const int* __restrict__ ei,
                const float* __restrict__ asrc,
                const float* __restrict__ adst,
                const float* __restrict__ m,
                const float* __restrict__ inv,
                const float* __restrict__ h,
                float* __restrict__ out) {
  int gw = (int)((blockIdx.x * blockDim.x + threadIdx.x) >> 5);
  int lane = threadIdx.x & 31;
  if (gw >= ETOT) return;
  int s, d; edge_nodes(ei, gw, s, d);
  int c = lane * 8;            // 32 lanes x 8 = 256 channels
  int head = c >> 6;
  int d4 = d * 4 + head;
  float v = asrc[s * 4 + head] + adst[d4];
  v = (v >= 0.0f) ? v : NEG_SLOPE * v;
  float alpha = expf(v - m[d4]) * inv[d4];
  const float4* hp = (const float4*)(h + (long)s * HC + c);
  float* op = out + (long)d * HC + c;
  float4 h0 = hp[0], h1 = hp[1];
  atomicAdd(op + 0, alpha * h0.x);
  atomicAdd(op + 1, alpha * h0.y);
  atomicAdd(op + 2, alpha * h0.z);
  atomicAdd(op + 3, alpha * h0.w);
  atomicAdd(op + 4, alpha * h1.x);
  atomicAdd(op + 5, alpha * h1.y);
  atomicAdd(op + 6, alpha * h1.z);
  atomicAdd(op + 7, alpha * h1.w);
}

// ------------------------------------------------------- bias + PReLU, in place
__global__ void finalize_kernel(float* __restrict__ out,
                                const float* __restrict__ bias,
                                const float* __restrict__ prelu_a) {
  int i = blockIdx.x * blockDim.x + threadIdx.x;
  if (i >= NNODES * HC) return;
  float v = out[i] + bias[i & (HC - 1)];
  float a = prelu_a[0];
  out[i] = (v >= 0.0f) ? v : a * v;
}

extern "C" void kernel_launch(void* const* d_in, const int* in_sizes, int n_in,
                              void* d_out, int out_size, void* d_ws, size_t ws_size,
                              hipStream_t stream) {
  (void)in_sizes; (void)n_in; (void)out_size; (void)ws_size;
  const float* x        = (const float*)d_in[0];
  const int*   ei       = (const int*)  d_in[1];
  const float* W        = (const float*)d_in[2];
  const float* att_src  = (const float*)d_in[3];
  const float* att_dst  = (const float*)d_in[4];
  const float* bias     = (const float*)d_in[5];
  const float* prelu_a  = (const float*)d_in[6];
  float* out = (float*)d_out;

  char* ws = (char*)d_ws;
  size_t off = 0;
  float*          hbuf  = (float*)(ws + off);         off += (size_t)NNODES * HC * 4;
  float*          asrc  = (float*)(ws + off);         off += (size_t)NNODES * NHEADS * 4;
  float*          adst  = (float*)(ws + off);         off += (size_t)NNODES * NHEADS * 4;
  unsigned int*   mkey  = (unsigned int*)(ws + off);
  float*          mflt  = (float*)(ws + off);         off += (size_t)NNODES * NHEADS * 4;
  float*          denom = (float*)(ws + off);         off += (size_t)NNODES * NHEADS * 4;
  unsigned short* wt_hi = (unsigned short*)(ws + off); off += (size_t)FIN * HC * 2;
  unsigned short* wt_lo = (unsigned short*)(ws + off); off += (size_t)FIN * HC * 2;

  const int T = 256;
  // zero accumulators (graph-safe, deterministic each call)
  zero_kernel<<<(NNODES * HC + T - 1) / T, T, 0, stream>>>(out, NNODES * HC);
  zero_kernel<<<(NNODES * NHEADS + T - 1) / T, T, 0, stream>>>((float*)mkey, NNODES * NHEADS);
  zero_kernel<<<(NNODES * NHEADS + T - 1) / T, T, 0, stream>>>(denom, NNODES * NHEADS);

  wprep_kernel<<<(FIN * HC + T - 1) / T, T, 0, stream>>>(W, wt_hi, wt_lo);

  gemm_bf16x3_kernel<<<(NNODES + 31) / 32, T, 0, stream>>>(x, wt_hi, wt_lo, hbuf);

  att_dot_kernel<<<(NNODES * NHEADS + T - 1) / T, T, 0, stream>>>(
      hbuf, att_src, att_dst, asrc, adst);

  edge_max_kernel<<<(ETOT * NHEADS + T - 1) / T, T, 0, stream>>>(ei, asrc, adst, mkey);
  decode_max_kernel<<<(NNODES * NHEADS + T - 1) / T, T, 0, stream>>>(mkey, mflt);

  edge_den_kernel<<<(ETOT * NHEADS + T - 1) / T, T, 0, stream>>>(ei, asrc, adst, mflt, denom);
  inv_den_kernel<<<(NNODES * NHEADS + T - 1) / T, T, 0, stream>>>(denom);

  edge_agg_kernel<<<(ETOT + 7) / 8, T, 0, stream>>>(ei, asrc, adst, mflt, denom, hbuf, out);

  finalize_kernel<<<(NNODES * HC + T - 1) / T, T, 0, stream>>>(out, bias, prelu_a);
}
